// ThinConv2d_32787780337775
// MI455X (gfx1250) — matrix operand order repaired and checked by reading, not compiled, not run on hardware
//
#include <hip/hip_runtime.h>
#include <hip/hip_bf16.h>
#include <stddef.h>

// ---- problem constants ----
#define BATCH 32
#define CIN   256
#define COUT  512
#define HH    56
#define WW    56
#define PIX   (HH * WW)          // 3136 pixels per image
#define KW    3

typedef float  v8f   __attribute__((ext_vector_type(8)));
typedef __bf16 v16bf __attribute__((ext_vector_type(16)));
typedef __bf16 v8bf  __attribute__((ext_vector_type(8)));

__device__ __forceinline__ unsigned short f32_to_bf16_rne(float f) {
    unsigned int u = __float_as_uint(f);
    u += 0x7FFFu + ((u >> 16) & 1u);   // round-to-nearest-even
    return (unsigned short)(u >> 16);
}

__device__ __forceinline__ v16bf cat8(v8bf lo, v8bf hi) {
    return __builtin_shufflevector(lo, hi,
           0, 1, 2, 3, 4, 5, 6, 7, 8, 9, 10, 11, 12, 13, 14, 15);
}

// ---------------------------------------------------------------------------
// Kernel 0: convert pointwise weights [COUT, CIN] f32 -> bf16 (row-major)
// ---------------------------------------------------------------------------
__global__ void cvt_w_bf16(const float* __restrict__ src,
                           unsigned short* __restrict__ dst, int n) {
    int i = blockIdx.x * blockDim.x + threadIdx.x;
    if (i < n) dst[i] = f32_to_bf16_rne(src[i]);
}

// ---------------------------------------------------------------------------
// Kernel 1: depthwise 3x3 (SAME, zero pad) + bias + ReLU, store bf16
// TRANSPOSED as y_t[b][pixel][c] (channel-contiguous per pixel).
// ---------------------------------------------------------------------------
__global__ __launch_bounds__(256)
void dw_relu_bf16(const float* __restrict__ x,
                  const float* __restrict__ dw_w,
                  const float* __restrict__ dw_b,
                  unsigned short* __restrict__ yt) {
    const int tx = threadIdx.x & 15;
    const int ty = threadIdx.x >> 4;
    const int w0 = blockIdx.x * 16;
    const int h0 = blockIdx.y * 16;
    const int bc = blockIdx.z;            // b*CIN + c
    const int c  = bc & (CIN - 1);
    const int b  = bc >> 8;

    __shared__ float sh[18][18];

    const float* xp = x + (size_t)bc * PIX;
    for (int i = threadIdx.x; i < 18 * 18; i += 256) {
        int r  = i / 18, q = i % 18;
        int gh = h0 + r - 1, gw = w0 + q - 1;
        float v = 0.0f;
        if (gh >= 0 && gh < HH && gw >= 0 && gw < WW) v = xp[gh * WW + gw];
        sh[r][q] = v;
    }
    __syncthreads();

    const int h = h0 + ty, w = w0 + tx;
    if (h < HH && w < WW) {
        const float* wp = dw_w + c * (KW * KW);
        float acc = dw_b[c];
#pragma unroll
        for (int ky = 0; ky < KW; ++ky)
#pragma unroll
            for (int kx = 0; kx < KW; ++kx)
                acc = fmaf(sh[ty + ky][tx + kx], wp[ky * KW + kx], acc);
        acc = fmaxf(acc, 0.0f);
        yt[((size_t)b * PIX + h * WW + w) * CIN + c] = f32_to_bf16_rne(acc);
    }
}

// ---------------------------------------------------------------------------
// Kernel 2: pointwise GEMM with v_wmma_f32_16x16x32_bf16.
//
// Workgroup (8 waves): 256 COUT rows x 64 pixels. B-tile (64 px x 32 ch,
// 4 KB/K-step) staged memory->LDS with GLOBAL_LOAD_ASYNC_TO_LDS_B128
// (ASYNCcnt path, no VGPR round-trip), double-buffered, padded row stride
// 40 elements (80 B) for a conflict-free ds_load_b128 bank permutation.
// Each wave: 2 M-tiles x 4 N-subtiles = 8 WMMAs per K-step, K = 256 in 8
// steps -> 64 WMMAs/wave. Accumulators seeded with pw_b.
// ---------------------------------------------------------------------------
#define NT      (PIX / 64)      // 49 pixel-groups per image
#define LROW    40              // padded LDS row stride (elements)
#define LBUF    (64 * LROW * 2) // bytes per LDS buffer (5120)

__global__ __launch_bounds__(256)
void pw_gemm_wmma(const unsigned short* __restrict__ wbf,   // [COUT][CIN] bf16
                  const unsigned short* __restrict__ yt,    // [B][PIX][CIN] bf16
                  const float* __restrict__ pw_b,
                  float* __restrict__ out) {
    const int tid  = threadIdx.x;
    const int wave = tid >> 5;
    const int lane = tid & 31;
    const int half = lane >> 4;
    const int l    = lane & 15;

    // Decode workgroup: b, half-of-COUT (wgm), pixel group (nt)
    const int wg  = blockIdx.x;
    const int b   = wg / (2 * NT);
    const int rem = wg % (2 * NT);
    const int wgm = rem / NT;            // 0/1: rows 0-255 / 256-511
    const int n0  = (rem % NT) * 64;

    const int m0 = (wgm * 16 + wave) * 16;        // first M-tile of this wave
    const int m1 = m0 + 128;                      // second (wave+8)

    __shared__ unsigned short sb[2][64 * LROW];   // 10 KB double buffer

    // Staging map: thread -> (pixel sp, 8-channel chunk sj); one async b128
    // per thread per K-step moves the 4 KB tile straight into LDS.
    const int sp = tid & 63;
    const int sj = (tid >> 6) * 8;                // 0,8,16,24
    const unsigned short* stage_src =
        yt + ((size_t)b * PIX + n0 + sp) * CIN + sj;
    // Low 32 bits of the flat shared address == workgroup-relative LDS offset.
    const unsigned lds0 =
        (unsigned)(size_t)(&sb[0][0] + sp * LROW + sj);

    // Accumulators seeded with bias (C/D layout: VGPR r -> M = r + half*8)
    v8f acc0[4], acc1[4];
#pragma unroll
    for (int r = 0; r < 8; ++r) {
        float b0 = pw_b[m0 + half * 8 + r];
        float b1 = pw_b[m1 + half * 8 + r];
#pragma unroll
        for (int t = 0; t < 4; ++t) { acc0[t][r] = b0; acc1[t][r] = b1; }
    }

    // A-fragment lane bases (16-bit A 16x32 layout: two 8-elem runs)
    const __bf16* A0 = (const __bf16*)wbf + (size_t)(m0 + l) * CIN + half * 8;
    const __bf16* A1 = (const __bf16*)wbf + (size_t)(m1 + l) * CIN + half * 8;

    // Prologue: async-stage K-step 0 into buffer 0
    asm volatile("global_load_async_to_lds_b128 %0, %1, off"
                 :: "v"(lds0), "v"(stage_src) : "memory");
    asm volatile("s_wait_asynccnt 0x0" ::: "memory");
    __syncthreads();

#pragma unroll
    for (int s = 0; s < 8; ++s) {
        const int kk  = s * 32;
        const int cur = s & 1;

        // Kick off next K-step's memory->LDS async copy into the other buffer
        // (previous barrier guarantees nobody still reads that buffer).
        if (s < 7) {
            const unsigned ldsn = lds0 + (cur ^ 1) * LBUF;
            const unsigned short* g = stage_src + kk + 32;
            asm volatile("global_load_async_to_lds_b128 %0, %1, off"
                         :: "v"(ldsn), "v"(g) : "memory");
        }

        // B-fragments from LDS: lane l, col n = t*16+l, K = half*8..+7, 16+half*8..+7
        v16bf bf[4];
#pragma unroll
        for (int t = 0; t < 4; ++t) {
            const unsigned short* p = &sb[cur][0] + (t * 16 + l) * LROW + half * 8;
            bf[t] = cat8(*(const v8bf*)p, *(const v8bf*)(p + 16));
        }

        // A-fragments from global (weights are WGP$/L2 resident)
        v16bf af0 = cat8(*(const v8bf*)(A0 + kk), *(const v8bf*)(A0 + kk + 16));
        v16bf af1 = cat8(*(const v8bf*)(A1 + kk), *(const v8bf*)(A1 + kk + 16));

#pragma unroll
        for (int t = 0; t < 4; ++t) {
            acc0[t] = __builtin_amdgcn_wmma_f32_16x16x32_bf16(
                false, af0, false, bf[t], (short)0, acc0[t], false, false);
            acc1[t] = __builtin_amdgcn_wmma_f32_16x16x32_bf16(
                false, af1, false, bf[t], (short)0, acc1[t], false, false);
        }

        // Ensure the async fill landed in LDS, then one barrier per step.
        if (s < 7) asm volatile("s_wait_asynccnt 0x0" ::: "memory");
        __syncthreads();
    }

    // Store D: VGPR r -> row m + half*8 + r, col n0 + t*16 + l
    float* op = out + (size_t)b * COUT * PIX;
#pragma unroll
    for (int r = 0; r < 8; ++r) {
        float* r0 = op + (size_t)(m0 + half * 8 + r) * PIX + n0 + l;
        float* r1 = op + (size_t)(m1 + half * 8 + r) * PIX + n0 + l;
#pragma unroll
        for (int t = 0; t < 4; ++t) {
            r0[t * 16] = acc0[t][r];
            r1[t * 16] = acc1[t][r];
        }
    }
}

// ---------------------------------------------------------------------------
extern "C" void kernel_launch(void* const* d_in, const int* in_sizes, int n_in,
                              void* d_out, int out_size, void* d_ws, size_t ws_size,
                              hipStream_t stream) {
    const float* x    = (const float*)d_in[0];   // [B, CIN, H, W]
    const float* dw_w = (const float*)d_in[1];   // [CIN, 1, 3, 3]
    const float* dw_b = (const float*)d_in[2];   // [CIN]
    const float* pw_w = (const float*)d_in[3];   // [COUT, CIN]
    const float* pw_b = (const float*)d_in[4];   // [COUT]
    float* out = (float*)d_out;                  // [B, COUT, H, W]

    // Workspace: bf16 weights (256 KiB), then bf16 transposed activations.
    unsigned short* wbf = (unsigned short*)d_ws;
    unsigned short* yt  = wbf + (size_t)COUT * CIN;
    (void)in_sizes; (void)n_in; (void)out_size; (void)ws_size;

    // 0) weights f32 -> bf16
    {
        int n = COUT * CIN;
        cvt_w_bf16<<<(n + 255) / 256, 256, 0, stream>>>(pw_w, wbf, n);
    }
    // 1) depthwise + bias + relu -> bf16 transposed
    {
        dim3 grid((WW + 15) / 16, (HH + 15) / 16, BATCH * CIN);
        dw_relu_bf16<<<grid, 256, 0, stream>>>(x, dw_w, dw_b, yt);
    }
    // 2) pointwise GEMM via WMMA bf16, async-LDS-staged B operand
    {
        int blocks = BATCH * 2 * NT;             // 32*2*49 = 3136 workgroups
        pw_gemm_wmma<<<blocks, 256, 0, stream>>>(wbf, yt, pw_b, out);
    }
}
